// MMSEMamba_12799002542240
// MI455X (gfx1250) — compile-verified, matrix operand order checked by reading, not run on hardware
//
#include <hip/hip_runtime.h>
#include <hip/hip_bf16.h>
#include <math.h>

typedef _Float16 h16;
typedef __attribute__((ext_vector_type(16))) _Float16 v16h;
typedef __attribute__((ext_vector_type(8)))  _Float16 v8h;
typedef __attribute__((ext_vector_type(8)))  float    v8f;
typedef __attribute__((ext_vector_type(4)))  unsigned int u32x4;
typedef __attribute__((ext_vector_type(8)))  int i32x8;
typedef __attribute__((ext_vector_type(4)))  int i32x4;

#define EPSF 1e-5f
#define BNF 0.999995000037500f   /* 1/sqrt(1+1e-5) */
#define NEG_INF (-3.402823466e38f)

static inline int cdiv(long a, long b) { return (int)((a + b - 1) / b); }

__device__ inline v16h pack16(v8h lo, v8h hi) {
  v16h r;
#pragma unroll
  for (int e = 0; e < 8; ++e) { r[e] = lo[e]; r[8 + e] = hi[e]; }
  return r;
}

// ---------------------------------------------------------------- utilities
__global__ void k_zero_f32(float* p, int n) {
  int i = blockIdx.x * 256 + threadIdx.x;
  if (i < n) p[i] = 0.f;
}
__global__ void k_zero_f16(h16* p, int n) {
  int i = blockIdx.x * 256 + threadIdx.x;
  if (i < n) p[i] = (h16)0.f;
}
__global__ void k_f32h(const float* __restrict__ s, h16* __restrict__ d, int n) {
  int i = blockIdx.x * 256 + threadIdx.x;
  if (i < n) d[i] = (h16)s[i];
}
// pad m_dt_w (2*512,16) -> (2*512,32) f16, zeros in cols 16..31
__global__ void k_cvt_dtw(const float* __restrict__ s, h16* __restrict__ d) {
  int i = blockIdx.x * 256 + threadIdx.x;
  if (i >= 2 * 512 * 32) return;
  int r = i & 31, dd = i >> 5;
  d[i] = (r < 16) ? (h16)s[dd * 16 + r] : (h16)0.f;
}
// dth[m*32+r] = dbl[m*48+r], r<16
__global__ void k_cvt_dth(const float* __restrict__ dbl, h16* __restrict__ dth) {
  int i = blockIdx.x * 256 + threadIdx.x;
  if (i >= 1024 * 16) return;
  int r = i & 15, m = i >> 4;
  dth[m * 32 + r] = (h16)dbl[m * 48 + r];
}

// ---------------------------------------------------------------- WMMA GEMM (1x1 tile; generic N multiple of 16)
// out[M,N] = A[M,K](row-major f16) x W[N,K]^T(row-major f16); optional bias[n], softplus
__global__ void k_gemm_rr1(const h16* __restrict__ A, int lda,
                           const h16* __restrict__ Bw, int ldb,
                           int M, int N, int K,
                           float* __restrict__ out, int ldo,
                           const float* __restrict__ bias, int sp) {
  int wv = (blockIdx.x * blockDim.x + threadIdx.x) >> 5;
  int lane = threadIdx.x & 31;
  int ntn = N >> 4;
  int total = (M >> 4) * ntn;
  if (wv >= total) return;
  int mt = wv / ntn, nt = wv % ntn;
  int row = lane & 15, half = lane >> 4;
  const h16* Ap = A  + (size_t)(mt * 16 + row) * lda + 8 * half;
  const h16* Bp = Bw + (size_t)(nt * 16 + row) * ldb + 8 * half;
  v8f acc = {};
  for (int k = 0; k < K; k += 32) {
    v16h a = pack16(*(const v8h*)(Ap + k), *(const v8h*)(Ap + k + 16));
    v16h b = pack16(*(const v8h*)(Bp + k), *(const v8h*)(Bp + k + 16));
    acc = __builtin_amdgcn_wmma_f32_16x16x32_f16(false, a, false, b, (short)0, acc, false, false);
  }
  int gn = nt * 16 + row;
  float bv = bias ? bias[gn] : 0.f;
  int gm0 = mt * 16 + 8 * half;
#pragma unroll
  for (int r = 0; r < 8; ++r) {
    float v = acc[r] + bv;
    if (sp) v = (v > 20.f) ? v : log1pf(expf(v));
    out[(size_t)(gm0 + r) * ldo + gn] = v;
  }
}

// ---------------------------------------------------------------- WMMA GEMM (2x2 tiles/wave; M%32==0, N%32==0)
__global__ void k_gemm_rr2(const h16* __restrict__ A, int lda,
                           const h16* __restrict__ Bw, int ldb,
                           int M, int N, int K,
                           float* __restrict__ out, int ldo,
                           const float* __restrict__ bias, int sp) {
  int wv = (blockIdx.x * blockDim.x + threadIdx.x) >> 5;
  int lane = threadIdx.x & 31;
  int ntn = N >> 5;
  int total = (M >> 5) * ntn;
  if (wv >= total) return;
  int mt = wv / ntn, nt = wv % ntn;
  int row = lane & 15, half = lane >> 4;
  const h16* Ap0 = A + (size_t)(mt * 32 + row) * lda + 8 * half;
  const h16* Ap1 = Ap0 + (size_t)16 * lda;
  const h16* Bp0 = Bw + (size_t)(nt * 32 + row) * ldb + 8 * half;
  const h16* Bp1 = Bp0 + (size_t)16 * ldb;
  v8f a00 = {}, a01 = {}, a10 = {}, a11 = {};
  for (int k = 0; k < K; k += 32) {
    v16h av0 = pack16(*(const v8h*)(Ap0 + k), *(const v8h*)(Ap0 + k + 16));
    v16h av1 = pack16(*(const v8h*)(Ap1 + k), *(const v8h*)(Ap1 + k + 16));
    v16h bv0 = pack16(*(const v8h*)(Bp0 + k), *(const v8h*)(Bp0 + k + 16));
    v16h bv1 = pack16(*(const v8h*)(Bp1 + k), *(const v8h*)(Bp1 + k + 16));
    a00 = __builtin_amdgcn_wmma_f32_16x16x32_f16(false, av0, false, bv0, (short)0, a00, false, false);
    a01 = __builtin_amdgcn_wmma_f32_16x16x32_f16(false, av0, false, bv1, (short)0, a01, false, false);
    a10 = __builtin_amdgcn_wmma_f32_16x16x32_f16(false, av1, false, bv0, (short)0, a10, false, false);
    a11 = __builtin_amdgcn_wmma_f32_16x16x32_f16(false, av1, false, bv1, (short)0, a11, false, false);
  }
  int gn0 = nt * 32 + row, gn1 = gn0 + 16;
  int gm0 = mt * 32 + 8 * half, gm1 = gm0 + 16;
  float bv0 = bias ? bias[gn0] : 0.f;
  float bv1 = bias ? bias[gn1] : 0.f;
#pragma unroll
  for (int r = 0; r < 8; ++r) {
    float v00 = a00[r] + bv0, v01 = a01[r] + bv1, v10 = a10[r] + bv0, v11 = a11[r] + bv1;
    if (sp) {
      v00 = (v00 > 20.f) ? v00 : log1pf(expf(v00));
      v01 = (v01 > 20.f) ? v01 : log1pf(expf(v01));
      v10 = (v10 > 20.f) ? v10 : log1pf(expf(v10));
      v11 = (v11 > 20.f) ? v11 : log1pf(expf(v11));
    }
    out[(size_t)(gm0 + r) * ldo + gn0] = v00;
    out[(size_t)(gm0 + r) * ldo + gn1] = v01;
    out[(size_t)(gm1 + r) * ldo + gn0] = v10;
    out[(size_t)(gm1 + r) * ldo + gn1] = v11;
  }
}

// ---------------------------------------------------------------- pointwise conv via WMMA + TDM-staged B slab
// out[b,o,hw] = sum_c W[o,c]*act[b,c,hw]; affine per-o; opt clip(0,6). Requires O==256, Cin==256.
// Block = 128 thr (4 waves), one hw-tile. Wave 0 issues tensor_load_to_lds for the 16x256
// K-slab (2D D#: tile_dim0=16, tile_dim1=Cin, stride0=HW, data_size=2B, pad 1 DWORD per
// 8 DWORDs -> LDS row stride 36B, bank-conflict-free gathers). Each wave does 4 o-tiles.
#define PW_ROW 18   /* halfs per c-row in LDS: 16 data + 2 pad */
__global__ void k_gemm_pw(const h16* __restrict__ W, const h16* __restrict__ act,
                          int O, int Cin, int HW,
                          float* __restrict__ outf, h16* __restrict__ outh,
                          const float* __restrict__ g, const float* __restrict__ bsh,
                          int clip6) {
  __shared__ __align__(16) h16 lb[256 * PW_ROW + 8];
  int b = blockIdx.z;
  int hw0 = blockIdx.x << 4;
  const h16* actb = act + (size_t)b * Cin * HW;
  int wave = threadIdx.x >> 5;
  int lane = threadIdx.x & 31;
  int row = lane & 15, half = lane >> 4;
  int ot0 = wave << 2;                       // this wave's first o-tile (of 16)
  if (lane == 0) __builtin_prefetch(W + (size_t)(ot0 * 16) * Cin, 0, 3);
  if (wave == 0) {
    // ---- Tensor Data Mover: DMA the [Cin x 16] slab into LDS (D# per ISA 08 §8.3-8.5)
    unsigned long long ga = (unsigned long long)(uintptr_t)(actb + hw0);
    unsigned int ldsoff = (unsigned int)(uintptr_t)lb;   // low 32 bits = LDS byte offset
    u32x4 g0;
    g0[0] = 1u;                                          // count=1, user descriptor
    g0[1] = ldsoff;                                      // lds_addr
    g0[2] = (unsigned int)(ga & 0xFFFFFFFFu);            // global_addr[31:0]
    g0[3] = (unsigned int)((ga >> 32) & 0x1FFFFFFu) | (2u << 30);  // addr[56:32] | type=2
    i32x8 g1;
    g1[0] = (1 << 16) | (1 << 20) | (2 << 22);           // data_size=2B, pad_enable, pad every 8 DW by 1 DW
    g1[1] = (16 << 16);                                  // tensor_dim0 = 16 (low16)
    g1[2] = (Cin << 16);                                 // tensor_dim0 hi=0 | tensor_dim1 lo16 = Cin
    g1[3] = (16 << 16);                                  // tensor_dim1 hi=0 | tile_dim0 = 16
    g1[4] = Cin;                                         // tile_dim1 = Cin | tile_dim2 = 0
    g1[5] = HW;                                          // tensor_dim0_stride[31:0]
    g1[6] = 0;
    g1[7] = 0;
    i32x4 z4 = {0, 0, 0, 0};
    i32x8 z8 = {0, 0, 0, 0, 0, 0, 0, 0};
    __builtin_amdgcn_tensor_load_to_lds(g0, g1, z4, z4, z8, 0);
    __builtin_amdgcn_s_wait_tensorcnt(0);
  }
  __syncthreads();
  v8f acc[4];
#pragma unroll
  for (int t = 0; t < 4; ++t)
#pragma unroll
    for (int r = 0; r < 8; ++r) acc[t][r] = 0.f;
  for (int k = 0; k < Cin; k += 32) {
    int kb = k + 8 * half;
    v16h bb;
#pragma unroll
    for (int e = 0; e < 8; ++e) {
      bb[e]     = lb[(kb + e) * PW_ROW + row];
      bb[8 + e] = lb[(kb + 16 + e) * PW_ROW + row];
    }
#pragma unroll
    for (int t = 0; t < 4; ++t) {
      const h16* Ap = W + (size_t)((ot0 + t) * 16 + row) * Cin + 8 * half;
      v16h a = pack16(*(const v8h*)(Ap + k), *(const v8h*)(Ap + k + 16));
      acc[t] = __builtin_amdgcn_wmma_f32_16x16x32_f16(false, a, false, bb, (short)0, acc[t], false, false);
    }
  }
  int hw = hw0 + row;
  size_t ob = (size_t)b * O * HW;
#pragma unroll
  for (int t = 0; t < 4; ++t) {
    int o0 = (ot0 + t) * 16 + 8 * half;
#pragma unroll
    for (int r = 0; r < 8; ++r) {
      int o = o0 + r;
      float v = acc[t][r] * (g[o] * BNF) + bsh[o];
      if (clip6) v = fminf(fmaxf(v, 0.f), 6.f);
      if (outf) outf[ob + (size_t)o * HW + hw] = v;
      else      outh[ob + (size_t)o * HW + hw] = (h16)v;
    }
  }
}

// ---------------------------------------------------------------- upsample (dw transposed conv + bn1 + relu -> f16)
__global__ void k_upsample(const float* __restrict__ x, const float* __restrict__ dw,
                           const float* __restrict__ g, const float* __restrict__ bsh,
                           h16* __restrict__ zr) {
  size_t idx = (size_t)blockIdx.x * 256 + threadIdx.x;   // B*C*128*128 = 2^25
  int b = (int)(idx >> 22);
  int rem = (int)(idx & ((1u << 22) - 1));
  int c = rem >> 14;
  int p = rem & 16383;
  int i = p >> 7, j = p & 127;
  const float* xp = x + ((size_t)(b * 256 + c)) * 4096;
  const float* wp = dw + c * 9;
  int i2 = i >> 1, j2 = j >> 1;
#define XV(hh, ww) xp[(hh) * 64 + (ww)]
  float acc;
  if ((i & 1) == 0) {
    if ((j & 1) == 0) acc = wp[4] * XV(i2, j2);
    else {
      acc = wp[5] * XV(i2, j2);
      if (j2 + 1 < 64) acc += wp[3] * XV(i2, j2 + 1);
    }
  } else {
    if ((j & 1) == 0) {
      acc = wp[7] * XV(i2, j2);
      if (i2 + 1 < 64) acc += wp[1] * XV(i2 + 1, j2);
    } else {
      acc = wp[8] * XV(i2, j2);
      if (j2 + 1 < 64) acc += wp[6] * XV(i2, j2 + 1);
      if (i2 + 1 < 64) acc += wp[2] * XV(i2 + 1, j2);
      if (i2 + 1 < 64 && j2 + 1 < 64) acc += wp[0] * XV(i2 + 1, j2 + 1);
    }
  }
#undef XV
  float v = acc * (g[c] * BNF) + bsh[c];
  v = fmaxf(v, 0.f);
  zr[idx] = (h16)v;
}

// ---------------------------------------------------------------- PE reductions (max/min + arg over H and W) via LDS plane
__global__ void k_pe_reduce(const float* __restrict__ x,
                            float* __restrict__ mv, float* __restrict__ mi,
                            float* __restrict__ nv, float* __restrict__ ni) {
  int bc = blockIdx.x;                              // b*256+c
  const float* xp = x + (size_t)bc * 4096;
  __shared__ float pl[64 * 65];
  for (int i = threadIdx.x; i < 4096; i += 256) {
    int hh = i >> 6, ww = i & 63;
    pl[hh * 65 + ww] = xp[i];
  }
  __syncthreads();
  int t = threadIdx.x;
  if (t < 64) {                                     // col: reduce over h, l = w = t
    float bm = NEG_INF, bn_ = -NEG_INF; int am = 0, an = 0;
    for (int hh = 0; hh < 64; ++hh) {
      float v = pl[hh * 65 + t];
      if (v > bm) { bm = v; am = hh; }
      if (v < bn_) { bn_ = v; an = hh; }
    }
    size_t base = (size_t)bc * 128 + t;
    mv[base] = bm; mi[base] = (float)am; nv[base] = bn_; ni[base] = (float)an;
  } else if (t < 128) {                             // row: reduce over w, l = 64 + h
    int hh = t - 64;
    float bm = NEG_INF, bn_ = -NEG_INF; int am = 0, an = 0;
    for (int ww = 0; ww < 64; ++ww) {
      float v = pl[hh * 65 + ww];
      if (v > bm) { bm = v; am = ww; }
      if (v < bn_) { bn_ = v; an = ww; }
    }
    size_t base = (size_t)bc * 128 + 64 + hh;
    mv[base] = bm; mi[base] = (float)am; nv[base] = bn_; ni[base] = (float)an;
  }
}

// ---------------------------------------------------------------- PE branch: +pos +idx, LN over c -> f16 (normal + flipped)
__global__ void k_pe_ln(const float* __restrict__ rv, const float* __restrict__ ri,
                        const float* __restrict__ pos, const float* __restrict__ g,
                        const float* __restrict__ bsh,
                        h16* __restrict__ uh, h16* __restrict__ ufl) {
  int rowi = blockIdx.x;                            // b*128+l
  int b = rowi >> 7, l = rowi & 127;
  int c = threadIdx.x;                              // 256
  float val = rv[((size_t)(b * 256 + c)) * 128 + l];
  float idx = ri[((size_t)(b * 256 + c)) * 128 + l];
  int li = (l < 64) ? l : (l - 64);
  float coords = fmaxf(((float)li + 0.5f) * 0.25f - 0.5f, 0.f);
  int i0 = (int)floorf(coords);
  int i1 = (i0 + 1 < 15) ? (i0 + 1) : 15;
  float wg = coords - (float)i0;
  float p = pos[c * 16 + i0] * (1.f - wg) + pos[c * 16 + i1] * wg;
  float t = val + p + idx;
  __shared__ float sr[256];
  __shared__ float mean_s, var_s;
  sr[c] = t; __syncthreads();
  for (int s = 128; s > 0; s >>= 1) { if (c < s) sr[c] += sr[c + s]; __syncthreads(); }
  if (c == 0) mean_s = sr[0] * (1.f / 256.f);
  __syncthreads();
  float d = t - mean_s;
  sr[c] = d * d; __syncthreads();
  for (int s = 128; s > 0; s >>= 1) { if (c < s) sr[c] += sr[c + s]; __syncthreads(); }
  if (c == 0) var_s = sr[0] * (1.f / 256.f);
  __syncthreads();
  float v = d * rsqrtf(var_s + EPSF) * g[c] + bsh[c];
  uh[(size_t)rowi * 256 + c] = (h16)v;
  ufl[(size_t)(b * 128 + (127 - l)) * 256 + c] = (h16)v;
}

// ---------------------------------------------------------------- generic LN over 256, optional f32 out and f16 out (flip/col_off)
__global__ void k_ln(const float* __restrict__ in, int ldin,
                     const float* __restrict__ g, const float* __restrict__ bsh,
                     float* __restrict__ outf, int ldof,
                     h16* __restrict__ outh, int ldoh, int col_off, int flip) {
  int rowi = blockIdx.x;
  int c = threadIdx.x;
  float t = in[(size_t)rowi * ldin + c];
  __shared__ float sr[256];
  __shared__ float mean_s, var_s;
  sr[c] = t; __syncthreads();
  for (int s = 128; s > 0; s >>= 1) { if (c < s) sr[c] += sr[c + s]; __syncthreads(); }
  if (c == 0) mean_s = sr[0] * (1.f / 256.f);
  __syncthreads();
  float d = t - mean_s;
  sr[c] = d * d; __syncthreads();
  for (int s = 128; s > 0; s >>= 1) { if (c < s) sr[c] += sr[c + s]; __syncthreads(); }
  if (c == 0) var_s = sr[0] * (1.f / 256.f);
  __syncthreads();
  float v = d * rsqrtf(var_s + EPSF) * g[c] + bsh[c];
  if (outf) outf[(size_t)rowi * ldof + c] = v;
  if (outh) {
    int b = rowi >> 7, l = rowi & 127;
    int ro = flip ? (b * 128 + (127 - l)) : rowi;
    outh[(size_t)ro * ldoh + col_off + c] = (h16)v;
  }
}

// ---------------------------------------------------------------- mamba depthwise causal conv1d + silu -> f16
__global__ void k_conv1d(const float* __restrict__ xz, const float* __restrict__ cw,
                         const float* __restrict__ cb, h16* __restrict__ xch) {
  int rowi = blockIdx.x;                            // 0..1023
  int j = threadIdx.x;                              // 512
  int b = rowi >> 7, l = rowi & 127;
  float acc = cb[j];
#pragma unroll
  for (int k = 0; k < 4; ++k) {
    int l2 = l - 3 + k;
    if (l2 >= 0) acc += xz[((size_t)(b * 128 + l2)) * 1024 + j] * cw[j * 4 + k];
  }
  acc = acc / (1.f + __expf(-acc));
  xch[(size_t)rowi * 512 + j] = (h16)acc;
}

// ---------------------------------------------------------------- selective scan (+ D skip, silu(zg) gate) -> f16
__global__ void k_scan(const float* __restrict__ xz, const h16* __restrict__ xch,
                       const float* __restrict__ dbl, const float* __restrict__ dt,
                       const float* __restrict__ Alog, const float* __restrict__ Dp,
                       h16* __restrict__ yh) {
  int b = blockIdx.x;
  int j = threadIdx.x;                              // 512
  float A[16];
#pragma unroll
  for (int s = 0; s < 16; ++s) A[s] = -__expf(Alog[j * 16 + s]);
  float Dv = Dp[j];
  float h[16];
#pragma unroll
  for (int s = 0; s < 16; ++s) h[s] = 0.f;
  __shared__ float sB[16], sC[16];
  for (int l = 0; l < 128; ++l) {
    size_t row = (size_t)b * 128 + l;
    if (j < 16) sB[j] = dbl[row * 48 + 16 + j];
    else if (j < 32) sC[j - 16] = dbl[row * 48 + 32 + (j - 16)];
    __syncthreads();
    float dtv = dt[row * 512 + j];
    float xcv = (float)xch[row * 512 + j];
    float y = 0.f;
#pragma unroll
    for (int s = 0; s < 16; ++s) {
      h[s] = __expf(dtv * A[s]) * h[s] + dtv * sB[s] * xcv;
      y += h[s] * sC[s];
    }
    y += xcv * Dv;
    float zg = xz[row * 1024 + 512 + j];
    y *= zg / (1.f + __expf(-zg));
    yh[row * 512 + j] = (h16)y;
    __syncthreads();
  }
}

// ---------------------------------------------------------------- gate (res broadcast sum, hswish-like) + gn1 stats
__global__ void k_gate_stats(const h16* __restrict__ zh, const float* __restrict__ rmax,
                             const float* __restrict__ rmin,
                             float* __restrict__ s1, float* __restrict__ s2) {
  size_t idx = (size_t)blockIdx.x * 256 + threadIdx.x;  // 2^25 elems
  int b = (int)(idx >> 22);
  int rem = (int)(idx & ((1u << 22) - 1));
  int c = rem >> 14;
  int p = rem & 16383;
  int i = p >> 7, j = p & 127;
  float rv = rmax[((size_t)(b * 128 + i)) * 256 + c] + rmin[((size_t)(b * 128 + j)) * 256 + c];
  float val = fminf(fmaxf(rv + 3.f, 0.f), 6.f) * (1.f / 6.f) * (float)zh[idx];
  __shared__ float ss[256], s2s[256];
  ss[threadIdx.x] = val; s2s[threadIdx.x] = val * val;
  __syncthreads();
  for (int s = 128; s > 0; s >>= 1) {
    if (threadIdx.x < s) { ss[threadIdx.x] += ss[threadIdx.x + s]; s2s[threadIdx.x] += s2s[threadIdx.x + s]; }
    __syncthreads();
  }
  if (threadIdx.x == 0) {
    int gi = b * 32 + (c >> 3);
    atomicAdd(&s1[gi], ss[0]);
    atomicAdd(&s2[gi], s2s[0]);
  }
}

// gn1-normalize gated values at even positions only -> f16 (8,256,64,64)
__global__ void k_gate_norm_sub(const h16* __restrict__ zh, const float* __restrict__ rmax,
                                const float* __restrict__ rmin,
                                const float* __restrict__ s1, const float* __restrict__ s2,
                                const float* __restrict__ gg, const float* __restrict__ gb,
                                h16* __restrict__ out) {
  size_t idx = (size_t)blockIdx.x * 256 + threadIdx.x;  // 2^23 elems
  int b = (int)(idx >> 20);
  int rem = (int)(idx & ((1u << 20) - 1));
  int c = rem >> 12;
  int p = rem & 4095;
  int hh = p >> 6, ww = p & 63;
  int i = 2 * hh, j = 2 * ww;
  float rv = rmax[((size_t)(b * 128 + i)) * 256 + c] + rmin[((size_t)(b * 128 + j)) * 256 + c];
  float val = fminf(fmaxf(rv + 3.f, 0.f), 6.f) * (1.f / 6.f) *
              (float)zh[((size_t)(b * 256 + c)) * 16384 + i * 128 + j];
  int gi = b * 32 + (c >> 3);
  const float invN = 1.f / 131072.f;
  float m = s1[gi] * invN;
  float var = s2[gi] * invN - m * m;
  out[idx] = (h16)((val - m) * rsqrtf(var + EPSF) * gg[c] + gb[c]);
}

// ---------------------------------------------------------------- group-norm stats over (8,256,64,64) f32
__global__ void k_gn_stats(const float* __restrict__ in, float* __restrict__ s1, float* __restrict__ s2) {
  size_t idx = (size_t)blockIdx.x * 256 + threadIdx.x;  // 2^23 elems
  int b = (int)(idx >> 20);
  int rem = (int)(idx & ((1u << 20) - 1));
  int c = rem >> 12;
  float v = in[idx];
  __shared__ float ss[256], s2s[256];
  ss[threadIdx.x] = v; s2s[threadIdx.x] = v * v;
  __syncthreads();
  for (int s = 128; s > 0; s >>= 1) {
    if (threadIdx.x < s) { ss[threadIdx.x] += ss[threadIdx.x + s]; s2s[threadIdx.x] += s2s[threadIdx.x + s]; }
    __syncthreads();
  }
  if (threadIdx.x == 0) {
    int gi = b * 32 + (c >> 3);
    atomicAdd(&s1[gi], ss[0]);
    atomicAdd(&s2[gi], s2s[0]);
  }
}

// apply gn; optional residual add; optional next-stage stats
__global__ void k_gn_apply(const float* __restrict__ in, const float* __restrict__ s1,
                           const float* __restrict__ s2, const float* __restrict__ gg,
                           const float* __restrict__ gb, const float* __restrict__ add,
                           float* __restrict__ out, float* __restrict__ ns1, float* __restrict__ ns2) {
  size_t idx = (size_t)blockIdx.x * 256 + threadIdx.x;  // 2^23 elems
  int b = (int)(idx >> 20);
  int rem = (int)(idx & ((1u << 20) - 1));
  int c = rem >> 12;
  int gi = b * 32 + (c >> 3);
  const float invN = 1.f / 32768.f;
  float m = s1[gi] * invN;
  float var = s2[gi] * invN - m * m;
  float v = (in[idx] - m) * rsqrtf(var + EPSF) * gg[c] + gb[c];
  if (add) v += add[idx];
  out[idx] = v;
  if (ns1) {
    __shared__ float ss[256], s2s[256];
    ss[threadIdx.x] = v; s2s[threadIdx.x] = v * v;
    __syncthreads();
    for (int s = 128; s > 0; s >>= 1) {
      if (threadIdx.x < s) { ss[threadIdx.x] += ss[threadIdx.x + s]; s2s[threadIdx.x] += s2s[threadIdx.x + s]; }
      __syncthreads();
    }
    if (threadIdx.x == 0) {
      atomicAdd(&ns1[gi], ss[0]);
      atomicAdd(&ns2[gi], s2s[0]);
    }
  }
}

// ================================================================ launcher
extern "C" void kernel_launch(void* const* d_in, const int* in_sizes, int n_in,
                              void* d_out, int out_size, void* d_ws, size_t ws_size,
                              hipStream_t stream) {
  const float* x        = (const float*)d_in[0];
  const float* pos_emb  = (const float*)d_in[1];
  const float* ln_g     = (const float*)d_in[2];
  const float* ln_b     = (const float*)d_in[3];
  const float* m_in_w   = (const float*)d_in[4];
  const float* m_conv_w = (const float*)d_in[5];
  const float* m_conv_b = (const float*)d_in[6];
  const float* m_xproj_w= (const float*)d_in[7];
  const float* m_dt_w   = (const float*)d_in[8];
  const float* m_dt_b   = (const float*)d_in[9];
  const float* m_A_log  = (const float*)d_in[10];
  const float* m_D      = (const float*)d_in[11];
  const float* m_out_w  = (const float*)d_in[12];
  const float* proj_w   = (const float*)d_in[13];
  const float* proj_b   = (const float*)d_in[14];
  const float* up_dw_w  = (const float*)d_in[15];
  const float* up_bn1_g = (const float*)d_in[16];
  const float* up_bn1_b = (const float*)d_in[17];
  const float* up_pw_w  = (const float*)d_in[18];
  const float* up_bn2_g = (const float*)d_in[19];
  const float* up_bn2_b = (const float*)d_in[20];
  const float* down_pw_w= (const float*)d_in[21];
  const float* down_bn_g= (const float*)d_in[22];
  const float* down_bn_b= (const float*)d_in[23];
  const float* gn_g     = (const float*)d_in[24];
  const float* gn_b     = (const float*)d_in[25];
  float* out = (float*)d_out;

  // ---- workspace bump allocator
  char* wsp = (char*)d_ws;
  auto alloc = [&](size_t bytes) -> void* {
    void* p = (void*)wsp;
    wsp += (bytes + 255) & ~(size_t)255;
    return p;
  };
  h16*   zr_h   = (h16*)  alloc((size_t)8 * 256 * 16384 * 2);
  h16*   z_h    = (h16*)  alloc((size_t)8 * 256 * 16384 * 2);
  float* rmaxv  = (float*)alloc((size_t)8 * 256 * 128 * 4);
  float* rmaxi  = (float*)alloc((size_t)8 * 256 * 128 * 4);
  float* rminv  = (float*)alloc((size_t)8 * 256 * 128 * 4);
  float* rmini  = (float*)alloc((size_t)8 * 256 * 128 * 4);
  h16*   u_max  = (h16*)  alloc((size_t)1024 * 256 * 2);
  h16*   u_min  = (h16*)  alloc((size_t)1024 * 256 * 2);
  h16*   u_maxf = (h16*)  alloc((size_t)1024 * 256 * 2);
  h16*   u_minf = (h16*)  alloc((size_t)1024 * 256 * 2);
  h16*   inw_h  = (h16*)  alloc((size_t)2 * 1024 * 256 * 2);
  h16*   xprj_h = (h16*)  alloc((size_t)2 * 48 * 512 * 2);
  h16*   dtw_h  = (h16*)  alloc((size_t)2 * 512 * 32 * 2);
  h16*   outw_h = (h16*)  alloc((size_t)2 * 256 * 512 * 2);
  h16*   prjw_h = (h16*)  alloc((size_t)256 * 512 * 2);
  h16*   uppw_h = (h16*)  alloc((size_t)256 * 256 * 2);
  h16*   dnpw_h = (h16*)  alloc((size_t)256 * 256 * 2);
  float* xz     = (float*)alloc((size_t)1024 * 1024 * 4);
  h16*   xc_h   = (h16*)  alloc((size_t)1024 * 512 * 2);
  float* dbl    = (float*)alloc((size_t)1024 * 48 * 4);
  h16*   dth    = (h16*)  alloc((size_t)1024 * 32 * 2);
  float* dt     = (float*)alloc((size_t)1024 * 512 * 4);
  h16*   y_h    = (h16*)  alloc((size_t)1024 * 512 * 2);
  float* mo     = (float*)alloc((size_t)1024 * 256 * 4);
  h16*   cat_mx = (h16*)  alloc((size_t)1024 * 512 * 2);
  h16*   cat_mn = (h16*)  alloc((size_t)1024 * 512 * 2);
  float* pre_mx = (float*)alloc((size_t)1024 * 256 * 4);
  float* pre_mn = (float*)alloc((size_t)1024 * 256 * 4);
  float* res_mx = (float*)alloc((size_t)1024 * 256 * 4);
  float* res_mn = (float*)alloc((size_t)1024 * 256 * 4);
  float* stats  = (float*)alloc((size_t)6 * 256 * 4);
  h16*   s_h    = (h16*)  alloc((size_t)8 * 256 * 4096 * 2);
  float* t1     = (float*)alloc((size_t)8 * 256 * 4096 * 4);
  float* s1a = stats, *s1b = stats + 256, *s2a = stats + 512, *s2b = stats + 768,
       * s3a = stats + 1024, *s3b = stats + 1280;

  // ---- init (every call: deterministic)
  k_zero_f32<<<cdiv(1536, 256), 256, 0, stream>>>(stats, 1536);
  k_zero_f16<<<cdiv(1024 * 32, 256), 256, 0, stream>>>(dth, 1024 * 32);

  // ---- weight conversions f32 -> f16
  k_f32h<<<cdiv(2 * 1024 * 256, 256), 256, 0, stream>>>(m_in_w, inw_h, 2 * 1024 * 256);
  k_f32h<<<cdiv(2 * 48 * 512, 256), 256, 0, stream>>>(m_xproj_w, xprj_h, 2 * 48 * 512);
  k_f32h<<<cdiv(2 * 256 * 512, 256), 256, 0, stream>>>(m_out_w, outw_h, 2 * 256 * 512);
  k_f32h<<<cdiv(256 * 512, 256), 256, 0, stream>>>(proj_w, prjw_h, 256 * 512);
  k_f32h<<<cdiv(256 * 256, 256), 256, 0, stream>>>(up_pw_w, uppw_h, 256 * 256);
  k_f32h<<<cdiv(256 * 256, 256), 256, 0, stream>>>(down_pw_w, dnpw_h, 256 * 256);
  k_cvt_dtw<<<cdiv(2 * 512 * 32, 256), 256, 0, stream>>>(m_dt_w, dtw_h);

  // ---- upsample branch: dw transposed conv + bn1 + relu, then pw conv (WMMA + TDM) + bn2 + clip
  k_upsample<<<cdiv((long)8 * 256 * 128 * 128, 256), 256, 0, stream>>>(x, up_dw_w, up_bn1_g, up_bn1_b, zr_h);
  k_gemm_pw<<<dim3(1024, 1, 8), 128, 0, stream>>>(uppw_h, zr_h, 256, 256, 16384,
                                                  nullptr, z_h, up_bn2_g, up_bn2_b, 1);

  // ---- PE reductions + LN
  k_pe_reduce<<<2048, 256, 0, stream>>>(x, rmaxv, rmaxi, rminv, rmini);
  k_pe_ln<<<1024, 256, 0, stream>>>(rmaxv, rmaxi, pos_emb, ln_g, ln_b, u_max, u_maxf);
  k_pe_ln<<<1024, 256, 0, stream>>>(rminv, rmini, pos_emb, ln_g, ln_b, u_min, u_minf);

  // ---- 4 mamba passes
  const h16* useq[4] = {u_max, u_min, u_maxf, u_minf};
  for (int run = 0; run < 4; ++run) {
    int d = (run < 2) ? 0 : 1;
    k_gemm_rr2<<<256, 128, 0, stream>>>(useq[run], 256, inw_h + (size_t)d * 1024 * 256, 256,
                                        1024, 1024, 256, xz, 1024, nullptr, 0);
    k_conv1d<<<1024, 512, 0, stream>>>(xz, m_conv_w + d * 512 * 4, m_conv_b + d * 512, xc_h);
    k_gemm_rr1<<<48, 128, 0, stream>>>(xc_h, 512, xprj_h + (size_t)d * 48 * 512, 512,
                                       1024, 48, 512, dbl, 48, nullptr, 0);
    k_cvt_dth<<<cdiv(1024 * 16, 256), 256, 0, stream>>>(dbl, dth);
    k_gemm_rr2<<<128, 128, 0, stream>>>(dth, 32, dtw_h + (size_t)d * 512 * 32, 32,
                                        1024, 512, 32, dt, 512, m_dt_b + d * 512, 1);
    k_scan<<<8, 512, 0, stream>>>(xz, xc_h, dbl, dt, m_A_log + d * 512 * 16, m_D + d * 512, y_h);
    k_gemm_rr2<<<64, 128, 0, stream>>>(y_h, 512, outw_h + (size_t)d * 256 * 512, 512,
                                       1024, 256, 512, mo, 256, nullptr, 0);
    h16* cat = (run == 0 || run == 2) ? cat_mx : cat_mn;
    int coff = (run < 2) ? 0 : 256;
    int flip = (run >= 2) ? 1 : 0;
    k_ln<<<1024, 256, 0, stream>>>(mo, 256, ln_g, ln_b, nullptr, 0, cat, 512, coff, flip);
  }

  // ---- final projection + LN -> res_max / res_min
  k_gemm_rr2<<<64, 128, 0, stream>>>(cat_mx, 512, prjw_h, 512, 1024, 256, 512, pre_mx, 256, proj_b, 0);
  k_gemm_rr2<<<64, 128, 0, stream>>>(cat_mn, 512, prjw_h, 512, 1024, 256, 512, pre_mn, 256, proj_b, 0);
  k_ln<<<1024, 256, 0, stream>>>(pre_mx, 256, ln_g, ln_b, res_mx, 256, nullptr, 0, 0, 0);
  k_ln<<<1024, 256, 0, stream>>>(pre_mn, 256, ln_g, ln_b, res_mn, 256, nullptr, 0, 0, 0);

  // ---- gate with z, gn1 stats, then normalize only the subsampled positions
  k_gate_stats<<<131072, 256, 0, stream>>>(z_h, res_mx, res_mn, s1a, s1b);
  k_gate_norm_sub<<<32768, 256, 0, stream>>>(z_h, res_mx, res_mn, s1a, s1b, gn_g, gn_b, s_h);

  // ---- down pointwise conv (WMMA + TDM) + bn
  k_gemm_pw<<<dim3(256, 1, 8), 128, 0, stream>>>(dnpw_h, s_h, 256, 256, 4096,
                                                 t1, nullptr, down_bn_g, down_bn_b, 0);

  // ---- gn2, +x, gn3 (in place on d_out)
  k_gn_stats<<<32768, 256, 0, stream>>>(t1, s2a, s2b);
  k_gn_apply<<<32768, 256, 0, stream>>>(t1, s2a, s2b, gn_g, gn_b, x, out, s3a, s3b);
  k_gn_apply<<<32768, 256, 0, stream>>>(out, s3a, s3b, gn_g, gn_b, nullptr, out, nullptr, nullptr);
}
